// StackGATs_88648124991108
// MI455X (gfx1250) — compile-verified, matrix operand order checked by reading, not run on hardware
//
#include <hip/hip_runtime.h>
#include <hip/hip_bf16.h>
#include <stdint.h>

// ---------------------------------------------------------------------------
// StackGATs on MI455X (gfx1250): 2-layer GAT.
// Dense X@W via v_wmma_f32_16x16x32_bf16 (bf16 in, f32 accum), fused logit
// matvecs. Sparse softmax/aggregate via streaming edge kernels + f32 atomics
// (bandwidth-bound: ~1.7 GB/layer -> the 23.3 TB/s HBM path dominates).
// ---------------------------------------------------------------------------

#define GAT_D 128          // feature dim (reference: D = 128)
#define KT 4               // K tiles of 32
#define NT 8               // N tiles of 16

typedef __attribute__((ext_vector_type(16))) __bf16 v16bf;
typedef __attribute__((ext_vector_type(8)))  float  v8f;

union FragU { uint4 u[2]; v16bf v; };

__device__ __forceinline__ uint32_t f32_to_bf16_bits(float f) {
    uint32_t u = __float_as_uint(f);
    // round-to-nearest-even
    return (u + 0x7FFFu + ((u >> 16) & 1u)) >> 16;
}

// ---------------------------------------------------------------------------
// fp32 -> bf16 (row-major, same layout)
// ---------------------------------------------------------------------------
__global__ void k_f32_to_bf16(const float* __restrict__ in,
                              uint16_t* __restrict__ out, int n) {
    int i = blockIdx.x * 256 + threadIdx.x;
    if (i < n) out[i] = (uint16_t)f32_to_bf16_bits(in[i]);
}

// ---------------------------------------------------------------------------
// Pack W (D x D fp32, row-major, K-major rows) into per-lane WMMA B-fragments:
// Wfrag[((kt*8+nt)*32 + lane)*8 + slot] = {W[k][n], W[k+1][n]} as 2xbf16.
//   n  = nt*16 + (lane&15)
//   kb = (lane>>4)*8
//   slot<4 : k = kt*32 + kb + 2*slot
//   slot>=4: k = kt*32 + 16 + kb + 2*(slot-4)
// Each lane's 16 bf16 are contiguous (32B) -> 2x ds_load_b128 in the GEMM.
// ---------------------------------------------------------------------------
__global__ void k_pack_W(const float* __restrict__ W, uint32_t* __restrict__ Wfrag) {
    int t = blockIdx.x * 256 + threadIdx.x;           // 4*8*32*8 = 8192 threads
    if (t >= KT * NT * 32 * 8) return;
    int slot = t & 7;
    int lane = (t >> 3) & 31;
    int nt   = (t >> 8) & 7;
    int kt   = t >> 11;
    int n  = nt * 16 + (lane & 15);
    int kb = (lane >> 4) * 8;
    int k  = kt * 32 + ((slot < 4) ? (kb + 2 * slot) : (16 + kb + 2 * (slot - 4)));
    uint32_t lo = f32_to_bf16_bits(W[(size_t)k * GAT_D + n]);
    uint32_t hi = f32_to_bf16_bits(W[(size_t)(k + 1) * GAT_D + n]);
    Wfrag[t] = lo | (hi << 16);
}

// ---------------------------------------------------------------------------
// GEMM: h = Xb(bf16) @ W, fused al_src = h@a_src, al_dst = h@a_dst.
// Block = 256 thr = 8 waves; wave w owns rows [blk*128 + w*16, +16) x all 128
// cols (8 accumulators of v8f). W fragments staged in 32KB LDS.
// ---------------------------------------------------------------------------
__global__ void __launch_bounds__(256)
k_gemm_al(const uint16_t* __restrict__ Xb, const uint32_t* __restrict__ Wfrag,
          const float* __restrict__ a_src, const float* __restrict__ a_dst,
          float* __restrict__ h, float* __restrict__ al_src,
          float* __restrict__ al_dst, int N) {
    __shared__ uint32_t ldsW[KT * NT * 32 * 8];       // 8192 dwords = 32 KB

    // cooperative copy of Wfrag into LDS (2048 uint4)
    {
        const uint4* ws = (const uint4*)Wfrag;
        uint4* wd = (uint4*)ldsW;
        #pragma unroll 2
        for (int i = threadIdx.x; i < 2048; i += 256) wd[i] = ws[i];
    }
    __syncthreads();

    const int wave  = threadIdx.x >> 5;
    const int lane  = threadIdx.x & 31;
    const int row16 = lane & 15;
    const int half  = lane >> 4;                       // 0 or 1
    const int blockm = blockIdx.x * 128 + wave * 16;   // wave's first row
    const int arow   = blockm + row16;                 // row this lane loads (A)
    const bool avalid = (arow < N);
    const int kboff = half * 8;                        // bf16 element offset

    v8f acc[NT] = {};

    #pragma unroll
    for (int kt = 0; kt < KT; ++kt) {
        FragU af;
        if (avalid) {
            const uint16_t* abase = Xb + (size_t)arow * GAT_D + kt * 32 + kboff;
            af.u[0] = *(const uint4*)(abase);          // K = base .. base+7
            af.u[1] = *(const uint4*)(abase + 16);     // K = base+16 .. base+23
        } else {
            af.u[0] = make_uint4(0u, 0u, 0u, 0u);
            af.u[1] = make_uint4(0u, 0u, 0u, 0u);
        }
        #pragma unroll
        for (int nt = 0; nt < NT; ++nt) {
            FragU bf;
            const uint4* bp =
                (const uint4*)(ldsW + ((size_t)(kt * NT + nt) * 32 + lane) * 8);
            bf.u[0] = bp[0];
            bf.u[1] = bp[1];
            acc[nt] = __builtin_amdgcn_wmma_f32_16x16x32_bf16(
                false, af.v, false, bf.v, (short)0, acc[nt], false, false);
        }
    }

    // Store h + fused attention-logit matvecs.
    // C layout: VGPR r, lanes 0-15 -> M = r, N = lane; lanes 16-31 -> M = r+8.
    #pragma unroll
    for (int r = 0; r < 8; ++r) {
        const int orow = blockm + r + 8 * half;
        float ss = 0.f, sd = 0.f;
        #pragma unroll
        for (int nt = 0; nt < NT; ++nt) {
            float v = acc[nt][r];
            int n = nt * 16 + row16;
            if (orow < N) h[(size_t)orow * GAT_D + n] = v;
            ss += v * a_src[n];
            sd += v * a_dst[n];
        }
        // reduce across the 16 lanes of this half-wave
        #pragma unroll
        for (int m2 = 8; m2 >= 1; m2 >>= 1) {
            ss += __shfl_xor(ss, m2, 16);
            sd += __shfl_xor(sd, m2, 16);
        }
        if (row16 == 0 && orow < N) {
            al_src[orow] = ss;
            al_dst[orow] = sd;
        }
    }
}

// ---------------------------------------------------------------------------
// init: out = 0, mmax = -inf, zsum = 0
// ---------------------------------------------------------------------------
__global__ void k_init(float* __restrict__ out, float* __restrict__ mmax,
                       float* __restrict__ zsum, int N) {
    int i = blockIdx.x * 256 + threadIdx.x;
    int nd = N * GAT_D;
    if (i < nd) out[i] = 0.f;
    if (i < N) {
        mmax[i] = -__builtin_inff();
        zsum[i] = 0.f;
    }
}

__device__ __forceinline__ void atomicMaxF32(float* addr, float v) {
    if (v >= 0.f)
        atomicMax((int*)addr, __float_as_int(v));
    else
        atomicMin((unsigned int*)addr, (unsigned int)__float_as_int(v));
}

// ---------------------------------------------------------------------------
// edge pass 1: e = leaky_relu(al_src[s] + al_dst[d]); segment max into mmax[d]
// edges [0,E) from edge_index, [E, E+N) are self loops
// ---------------------------------------------------------------------------
__global__ void k_edge_logit_max(const int* __restrict__ src,
                                 const int* __restrict__ dst,
                                 const float* __restrict__ als,
                                 const float* __restrict__ ald,
                                 float* __restrict__ eraw,
                                 float* __restrict__ mmax, int E, int N) {
    int i = blockIdx.x * 256 + threadIdx.x;
    if (i >= E + N) return;
    int s, d;
    if (i < E) { s = src[i]; d = dst[i]; } else { s = d = i - E; }
    float e = als[s] + ald[d];
    e = (e > 0.f) ? e : 0.2f * e;                     // leaky_relu, slope 0.2
    eraw[i] = e;
    atomicMaxF32(&mmax[d], e);
}

// ---------------------------------------------------------------------------
// edge pass 2: e = exp(e - mmax[d]); segment sum into zsum[d]
// ---------------------------------------------------------------------------
__global__ void k_edge_exp_sum(const int* __restrict__ dst,
                               float* __restrict__ eraw,
                               const float* __restrict__ mmax,
                               float* __restrict__ zsum, int E, int N) {
    int i = blockIdx.x * 256 + threadIdx.x;
    if (i >= E + N) return;
    int d = (i < E) ? dst[i] : (i - E);
    float e = __expf(eraw[i] - mmax[d]);
    eraw[i] = e;                                      // now unnormalized alpha
    atomicAdd(&zsum[d], e);
}

// ---------------------------------------------------------------------------
// edge pass 3: out[d] += (eraw[i]/zsum[d]) * h[s]. One wave per edge:
// 32 lanes x float4 = 128 features; f32 atomics land in L2 (out fits easily
// in the 192MB L2).
// ---------------------------------------------------------------------------
__global__ void __launch_bounds__(256)
k_edge_scatter(const int* __restrict__ src, const int* __restrict__ dst,
               const float* __restrict__ eraw, const float* __restrict__ zsum,
               const float* __restrict__ h, float* __restrict__ out,
               int E, int N) {
    int lane = threadIdx.x & 31;
    int widx = threadIdx.x >> 5;
    int edge = blockIdx.x * 8 + widx;
    if (edge >= E + N) return;
    int s, d;
    if (edge < E) { s = src[edge]; d = dst[edge]; } else { s = d = edge - E; }
    const float* hrow = h + (size_t)s * GAT_D;
    __builtin_prefetch(hrow + lane * 4, 0, 0);        // global_prefetch
    float alpha = eraw[edge] / zsum[d];
    float4 hv = ((const float4*)hrow)[lane];
    float* op = out + (size_t)d * GAT_D + lane * 4;
    atomicAdd(op + 0, hv.x * alpha);
    atomicAdd(op + 1, hv.y * alpha);
    atomicAdd(op + 2, hv.z * alpha);
    atomicAdd(op + 3, hv.w * alpha);
}

// ---------------------------------------------------------------------------
// out[n][d] += b[d]
// ---------------------------------------------------------------------------
__global__ void k_add_bias(float* __restrict__ out, const float* __restrict__ b,
                           int N) {
    int i = blockIdx.x * 256 + threadIdx.x;
    if (i < N * GAT_D) out[i] += b[i & (GAT_D - 1)];
}

// ---------------------------------------------------------------------------
// host side
// ---------------------------------------------------------------------------
static inline size_t align256(size_t x) { return (x + 255) & ~(size_t)255; }

extern "C" void kernel_launch(void* const* d_in, const int* in_sizes, int n_in,
                              void* d_out, int out_size, void* d_ws, size_t ws_size,
                              hipStream_t stream) {
    const float* x   = (const float*)d_in[0];
    const int*   ei  = (const int*)d_in[1];
    const float* W1  = (const float*)d_in[2];
    const float* as1 = (const float*)d_in[3];
    const float* ad1 = (const float*)d_in[4];
    const float* b1  = (const float*)d_in[5];
    const float* W2  = (const float*)d_in[6];
    const float* as2 = (const float*)d_in[7];
    const float* ad2 = (const float*)d_in[8];
    const float* b2  = (const float*)d_in[9];

    const int D = GAT_D;
    const int N = in_sizes[0] / D;
    const int E = in_sizes[1] / 2;
    const int* src = ei;
    const int* dst = ei + E;

    char* ws = (char*)d_ws;
    size_t off = 0;
    auto carve = [&](size_t bytes) -> char* {
        char* p = ws + off;
        off += align256(bytes);
        return p;
    };
    uint16_t* Xb    = (uint16_t*)carve((size_t)N * D * sizeof(uint16_t));
    uint32_t* Wf    = (uint32_t*)carve((size_t)KT * NT * 32 * 8 * sizeof(uint32_t));
    float*    h     = (float*)carve((size_t)N * D * sizeof(float));
    float*    als   = (float*)carve((size_t)N * sizeof(float));
    float*    ald   = (float*)carve((size_t)N * sizeof(float));
    float*    mmax  = (float*)carve((size_t)N * sizeof(float));
    float*    zsum  = (float*)carve((size_t)N * sizeof(float));
    float*    eraw  = (float*)carve((size_t)(E + N) * sizeof(float));
    float*    O1    = (float*)carve((size_t)N * D * sizeof(float));

    const int nd       = N * D;
    const int g_nd     = (nd + 255) / 256;
    const int g_gemm   = (N + 127) / 128;
    const int g_edge   = (E + N + 255) / 256;
    const int g_scat   = (E + N + 7) / 8;

    auto run_layer = [&](const float* xin, const float* W, const float* as,
                         const float* ad, const float* bb, float* outp) {
        k_f32_to_bf16<<<g_nd, 256, 0, stream>>>(xin, Xb, nd);
        k_pack_W<<<32, 256, 0, stream>>>(W, Wf);
        k_gemm_al<<<g_gemm, 256, 0, stream>>>(Xb, Wf, as, ad, h, als, ald, N);
        k_init<<<g_nd, 256, 0, stream>>>(outp, mmax, zsum, N);
        k_edge_logit_max<<<g_edge, 256, 0, stream>>>(src, dst, als, ald, eraw,
                                                     mmax, E, N);
        k_edge_exp_sum<<<g_edge, 256, 0, stream>>>(dst, eraw, mmax, zsum, E, N);
        k_edge_scatter<<<g_scat, 256, 0, stream>>>(src, dst, eraw, zsum, h,
                                                   outp, E, N);
        k_add_bias<<<g_nd, 256, 0, stream>>>(outp, bb, N);
    };

    run_layer(x,  W1, as1, ad1, b1, O1);
    run_layer(O1, W2, as2, ad2, b2, (float*)d_out);
}
